// LWMMambaBlock_20023137534123
// MI455X (gfx1250) — matrix operand order repaired and checked by reading, not compiled, not run on hardware
//
#include <hip/hip_runtime.h>

// ---------------------------------------------------------------------------
// Types for CDNA5 WMMA / TDM
// ---------------------------------------------------------------------------
typedef __attribute__((ext_vector_type(16))) __bf16 v16bf;
typedef __attribute__((ext_vector_type(8)))  float  v8f;
typedef __attribute__((ext_vector_type(4)))  unsigned int u32x4;
typedef __attribute__((ext_vector_type(8)))  int i32x8;
typedef __attribute__((ext_vector_type(4)))  int i32x4;

#ifndef __has_builtin
#define __has_builtin(x) 0
#endif
#if __has_builtin(__builtin_amdgcn_tensor_load_to_lds)
#define HAVE_TDM 1
#else
#define HAVE_TDM 0
#endif

struct Pack32 { uint4 lo, hi; };   // 32 bytes == v16bf

__device__ __forceinline__ unsigned short f32_to_bf16(float f) {
  unsigned int u = __float_as_uint(f);
  u += 0x7FFFu + ((u >> 16) & 1u);          // round-to-nearest-even
  return (unsigned short)(u >> 16);
}

// A-fragment (16-bit A 16x32): lane holds 16 contiguous K values starting at (lane>=16 ? 16 : 0)
__device__ __forceinline__ v16bf load_frag_a(const unsigned short* p) {
  Pack32 t;
  t.lo = *reinterpret_cast<const uint4*>(p);
  t.hi = *reinterpret_cast<const uint4*>(p + 8);
  return __builtin_bit_cast(v16bf, t);
}
// B-fragment: lane holds 16 contiguous K values starting at (lane>=16 ? 16 : 0)
__device__ __forceinline__ v16bf load_frag_b(const unsigned short* p) {
  Pack32 t;
  t.lo = *reinterpret_cast<const uint4*>(p);
  t.hi = *reinterpret_cast<const uint4*>(p + 8);
  return __builtin_bit_cast(v16bf, t);
}

// ---------------------------------------------------------------------------
// GEMM tiling constants
// ---------------------------------------------------------------------------
constexpr int KC     = 256;              // K-chunk staged to LDS per TDM op
constexpr int BROWS  = 64;               // N rows per block tile
constexpr int LDS_RS = KC + 8;           // padded row stride (528B) -> no bank conflicts
constexpr int LDS_EL = BROWS * LDS_RS;   // ushorts per buffer

#if HAVE_TDM
// TDM: async DMA of a [64 x KC] bf16 tile of BT[N,K] (row-major, K contiguous)
// into LDS with 4-DWORD padding every 512B row (pad_interval=6, pad_amount=3).
__device__ __forceinline__ void tdm_load_B(unsigned ldsOff, const unsigned short* g,
                                           unsigned K, unsigned N) {
  unsigned long long ga = (unsigned long long)(size_t)g;
  u32x4 g0;
  g0[0] = 1u;                                            // count=1, user mode
  g0[1] = ldsOff;                                        // lds_addr (bytes)
  g0[2] = (unsigned)ga;                                  // global_addr[31:0]
  g0[3] = (unsigned)((ga >> 32) & 0x01FFFFFFull)         // global_addr[56:32]
        | (2u << 30);                                    // type = 2 ("image")
  i32x8 g1;
  g1[0] = (int)((1u << 16)                               // data_size = 2 bytes
        | (1u << 20)                                     // pad_enable
        | (6u << 22)                                     // pad_interval: 128 DW (512B)
        | (3u << 25));                                   // pad_amount: 4 DW (16B)
  g1[1] = (int)((K & 0xFFFFu) << 16);                    // tensor_dim0[15:0]
  g1[2] = (int)((K >> 16) | ((N & 0xFFFFu) << 16));      // dim0 hi | tensor_dim1 lo
  g1[3] = (int)((N >> 16) | ((unsigned)KC << 16));       // dim1 hi | tile_dim0 = KC
  g1[4] = BROWS;                                         // tile_dim1 = 64, tile_dim2 = 0
  g1[5] = (int)K;                                        // tensor_dim0_stride lo32
  g1[6] = 0;                                             // stride hi | dim1_stride lo
  g1[7] = 0;
  i32x4 z4 = {0, 0, 0, 0};
#if __clang_major__ >= 23
  i32x8 z8 = {0, 0, 0, 0, 0, 0, 0, 0};
  __builtin_amdgcn_tensor_load_to_lds(g0, g1, z4, z4, z8, 0);
#else
  __builtin_amdgcn_tensor_load_to_lds(g0, g1, z4, z4, 0);
#endif
}
#endif

// ---------------------------------------------------------------------------
// f32 -> bf16 elementwise convert
// ---------------------------------------------------------------------------
__global__ void k_f32_to_bf16(const float* __restrict__ in,
                              unsigned short* __restrict__ out, int n) {
  int i = blockIdx.x * blockDim.x + threadIdx.x;
  int stride = gridDim.x * blockDim.x;
  for (; i < n; i += stride) out[i] = f32_to_bf16(in[i]);
}

// ---------------------------------------------------------------------------
// W[K,N] f32  ->  WT[N,K] bf16   (LDS-tiled, both sides coalesced)
// ---------------------------------------------------------------------------
__global__ void k_transpose_bf16(const float* __restrict__ W,
                                 unsigned short* __restrict__ WT,
                                 int K, int N) {
  __shared__ unsigned short tile[32][33];
  const int kBase = blockIdx.x * 32;
  const int nBase = blockIdx.y * 32;
  const int tx = threadIdx.x, ty = threadIdx.y;
#pragma unroll
  for (int i = 0; i < 32; i += 8)
    tile[ty + i][tx] = f32_to_bf16(W[(size_t)(kBase + ty + i) * N + nBase + tx]);
  __syncthreads();
#pragma unroll
  for (int i = 0; i < 32; i += 8)
    WT[(size_t)(nBase + ty + i) * K + kBase + tx] = tile[tx][ty + i];
}

// ---------------------------------------------------------------------------
// WMMA GEMM:  out = act(A[M,K]bf16 @ BT[N,K]bf16^T + bias[N])
// block = 256 threads (8 waves); block tile 256(M) x 64(N); wave tile 32x64.
// B staged to LDS by TDM (double-buffered, TENSORcnt), A streamed from global.
// act: 0 = none, 1 = exact GELU, 2 = SiLU
// ---------------------------------------------------------------------------
__global__ void __launch_bounds__(256)
k_wmma_gemm(const unsigned short* __restrict__ A,
            const unsigned short* __restrict__ BT,
            const float* __restrict__ bias,
            unsigned short* __restrict__ outBf,   // optional bf16 output
            float* __restrict__ outF,             // optional f32 output
            int Ntot, int K, int act) {
  __shared__ unsigned short ldsB[2][LDS_EL];

  const int lane = threadIdx.x & 31;
  const int wave = threadIdx.x >> 5;
  const int col  = lane & 15;
  const int hi   = lane >> 4;                 // 0 or 1
  const int mBase = blockIdx.x * 256 + wave * 32;
  const int nBase = blockIdx.y * 64;

  const unsigned short* aRow0 = A + (size_t)(mBase + col) * K + (hi << 4);
  const unsigned short* aRow1 = A + (size_t)(mBase + 16 + col) * K + (hi << 4);

  v8f acc[8];
#pragma unroll
  for (int i = 0; i < 8; ++i) acc[i] = v8f{};

#if HAVE_TDM
  // ---- preload chunk 0 via Tensor Data Mover ----
  if (wave == 0) {
    tdm_load_B((unsigned)(size_t)&ldsB[0][0],
               BT + (size_t)nBase * K, (unsigned)K, (unsigned)Ntot);
    __builtin_amdgcn_s_wait_tensorcnt(0);
  }
  __syncthreads();

  for (int kc = 0; kc < K; kc += KC) {
    const int  cur  = (kc / KC) & 1;
    const bool more = (kc + KC) < K;
    if (wave == 0 && more)                       // async: DMA next chunk now
      tdm_load_B((unsigned)(size_t)&ldsB[cur ^ 1][0],
                 BT + (size_t)nBase * K + kc + KC, (unsigned)K, (unsigned)Ntot);

    const unsigned short* bL = &ldsB[cur][0];
#pragma unroll
    for (int k = 0; k < KC; k += 32) {
      __builtin_prefetch(aRow0 + kc + k + 256, 0, 1);
      v16bf aF0 = load_frag_a(aRow0 + kc + k);
      v16bf aF1 = load_frag_a(aRow1 + kc + k);
#pragma unroll
      for (int j = 0; j < 4; ++j) {
        const unsigned short* bp = bL + (size_t)(j * 16 + col) * LDS_RS + k + (hi << 4);
        v16bf bF = load_frag_b(bp);
        acc[j]     = __builtin_amdgcn_wmma_f32_16x16x32_bf16(false, aF0, false, bF, (short)0, acc[j],     false, false);
        acc[j + 4] = __builtin_amdgcn_wmma_f32_16x16x32_bf16(false, aF1, false, bF, (short)0, acc[j + 4], false, false);
      }
    }
    if (wave == 0 && more) __builtin_amdgcn_s_wait_tensorcnt(0);
    __syncthreads();
  }
#else
  // ---- fallback: cooperative global->LDS staging (host pass / no TDM) ----
  for (int kc = 0; kc < K; kc += KC) {
    __syncthreads();
    for (int q = threadIdx.x; q < (BROWS * KC) / 8; q += 256) {
      const int row = q >> 5, cq = q & 31;   // 32 uint4 per 256-element row
      *reinterpret_cast<uint4*>(&ldsB[0][(size_t)row * LDS_RS + cq * 8]) =
        *reinterpret_cast<const uint4*>(BT + (size_t)(nBase + row) * K + kc + cq * 8);
    }
    __syncthreads();
    const unsigned short* bL = &ldsB[0][0];
#pragma unroll
    for (int k = 0; k < KC; k += 32) {
      v16bf aF0 = load_frag_a(aRow0 + kc + k);
      v16bf aF1 = load_frag_a(aRow1 + kc + k);
#pragma unroll
      for (int j = 0; j < 4; ++j) {
        const unsigned short* bp = bL + (size_t)(j * 16 + col) * LDS_RS + k + (hi << 4);
        v16bf bF = load_frag_b(bp);
        acc[j]     = __builtin_amdgcn_wmma_f32_16x16x32_bf16(false, aF0, false, bF, (short)0, acc[j],     false, false);
        acc[j + 4] = __builtin_amdgcn_wmma_f32_16x16x32_bf16(false, aF1, false, bF, (short)0, acc[j + 4], false, false);
      }
    }
  }
#endif

  // C/D layout (ISA 7.12.2): VGPR r -> M = r + 8*hi, N = lane&15
  const int mOff = hi << 3;
#pragma unroll
  for (int f = 0; f < 2; ++f) {
#pragma unroll
    for (int j = 0; j < 4; ++j) {
      const int n = nBase + j * 16 + col;
      const float bc = bias[n];
#pragma unroll
      for (int r = 0; r < 8; ++r) {
        float v = acc[f * 4 + j][r] + bc;
        if (act == 1)      v = 0.5f * v * (1.0f + erff(v * 0.70710678118654752f));
        else if (act == 2) v = v / (1.0f + __expf(-v));
        const size_t idx = (size_t)(mBase + f * 16 + mOff + r) * (size_t)Ntot + n;
        if (outBf) outBf[idx] = f32_to_bf16(v);
        if (outF)  outF[idx]  = v;
      }
    }
  }
}

// ---------------------------------------------------------------------------
// CB[d,n] = C[n,d] * min(softplus(dp[d]),10) * B[d,n]
// ---------------------------------------------------------------------------
__global__ void k_cb(const float* __restrict__ dp, const float* __restrict__ Bm,
                     const float* __restrict__ Cm, float* __restrict__ CB, int D) {
  int d = blockIdx.x * blockDim.x + threadIdx.x;
  if (d >= D) return;
  float x = dp[d];
  float sp = (x > 20.f) ? x : log1pf(expf(x));
  float delta = fminf(sp, 10.f);
#pragma unroll
  for (int n = 0; n < 16; ++n)
    CB[d * 16 + n] = Cm[(size_t)n * D + d] * delta * Bm[d * 16 + n];
}

// ---------------------------------------------------------------------------
// Fused: depthwise causal conv x2 -> 16-state SSM recurrence x2 ->
//        out = gate * (v1 + v2) + x.   One thread per (batch, channel).
// ---------------------------------------------------------------------------
__global__ void k_conv_ssm_out(const float* __restrict__ val,
                               const float* __restrict__ gate,
                               const float* __restrict__ x,
                               const float* __restrict__ cw1, const float* __restrict__ cb1,
                               const float* __restrict__ cw2, const float* __restrict__ cb2,
                               const float* __restrict__ A1, const float* __restrict__ CB1,
                               const float* __restrict__ A2, const float* __restrict__ CB2,
                               float* __restrict__ out,
                               int Bn, int L, int D) {
  int t = blockIdx.x * blockDim.x + threadIdx.x;
  if (t >= Bn * D) return;
  const int b = t / D, d = t % D;

  float w1[4], w2[4];
#pragma unroll
  for (int k = 0; k < 4; ++k) { w1[k] = cw1[d * 4 + k]; w2[k] = cw2[d * 4 + k]; }
  const float bb1 = cb1[d], bb2 = cb2[d];

  float a1[16], a2[16], c1v[16], c2v[16], h1[16], h2[16];
#pragma unroll
  for (int n = 0; n < 16; ++n) {
    a1[n] = A1[n * 17];  a2[n] = A2[n * 17];          // diag(A)
    c1v[n] = CB1[d * 16 + n];  c2v[n] = CB2[d * 16 + n];
    h1[n] = 0.f;  h2[n] = 0.f;
  }

  float vm3 = 0.f, vm2 = 0.f, vm1 = 0.f;              // causal left zero-pad
  const size_t base = (size_t)b * L * D + d;
  for (int l = 0; l < L; ++l) {
    const size_t gi = base + (size_t)l * D;
    const float xv = val[gi];
    float c1 = bb1 + w1[0] * vm3 + w1[1] * vm2 + w1[2] * vm1 + w1[3] * xv;
    float c2 = bb2 + w2[0] * vm3 + w2[1] * vm2 + w2[2] * vm1 + w2[3] * xv;
    vm3 = vm2; vm2 = vm1; vm1 = xv;

    float s1 = 0.f, s2 = 0.f;
#pragma unroll
    for (int n = 0; n < 16; ++n) {
      h1[n] = fmaf(a1[n], h1[n], c1);  s1 = fmaf(c1v[n], h1[n], s1);
      h2[n] = fmaf(a2[n], h2[n], c2);  s2 = fmaf(c2v[n], h2[n], s2);
    }
    const float vsum = (c1 + s1) + (c2 + s2);
    out[gi] = gate[gi] * vsum + x[gi];
  }
}

// ---------------------------------------------------------------------------
// Host-side orchestration
// ---------------------------------------------------------------------------
extern "C" void kernel_launch(void* const* d_in, const int* in_sizes, int n_in,
                              void* d_out, int out_size, void* d_ws, size_t ws_size,
                              hipStream_t stream) {
  (void)in_sizes; (void)n_in; (void)out_size; (void)ws_size;

  constexpr int D  = 1024, M = 4 * 2048;   // B*L = 8192
  constexpr int D4 = 4 * D;

  // workspace layout (all sizes multiples of 256B)
  char* ws = (char*)d_ws;
  unsigned short* wT   = (unsigned short*)ws;                 ws += (size_t)D4 * D * 2;  // 8 MiB
  unsigned short* xbf  = (unsigned short*)ws;                 ws += (size_t)M * D * 2;   // 16 MiB
  unsigned short* g0   = (unsigned short*)ws;                 ws += (size_t)M * D * 2;   // 16 MiB
  unsigned short* hbuf = (unsigned short*)ws;                 ws += (size_t)M * D4 * 2;  // 64 MiB
  float* gate = (float*)ws;                                   ws += (size_t)M * D * 4;   // 32 MiB
  float* valb = (float*)ws;                                   ws += (size_t)M * D * 4;   // 32 MiB
  float* CB1  = (float*)ws;                                   ws += (size_t)D * 16 * 4;
  float* CB2  = (float*)ws;                                   ws += (size_t)D * 16 * 4;

  const dim3 tB(32, 8);

  for (int br = 0; br < 2; ++br) {
    const float* xin = (const float*)d_in[br];
    const int p = 2 + br * 20;  // params: Wg bg W1 b1 W2 b2 Wv bv cw1 cb1 cw2 cb2 A1 d1 B1 C1 A2 d2 B2 C2
    const float* Wg = (const float*)d_in[p + 0];  const float* bg = (const float*)d_in[p + 1];
    const float* W1 = (const float*)d_in[p + 2];  const float* b1 = (const float*)d_in[p + 3];
    const float* W2 = (const float*)d_in[p + 4];  const float* b2 = (const float*)d_in[p + 5];
    const float* Wv = (const float*)d_in[p + 6];  const float* bv = (const float*)d_in[p + 7];
    const float* cw1 = (const float*)d_in[p + 8]; const float* cb1 = (const float*)d_in[p + 9];
    const float* cw2 = (const float*)d_in[p + 10];const float* cb2 = (const float*)d_in[p + 11];
    const float* A1 = (const float*)d_in[p + 12]; const float* d1 = (const float*)d_in[p + 13];
    const float* B1 = (const float*)d_in[p + 14]; const float* C1 = (const float*)d_in[p + 15];
    const float* A2 = (const float*)d_in[p + 16]; const float* d2 = (const float*)d_in[p + 17];
    const float* B2 = (const float*)d_in[p + 18]; const float* C2 = (const float*)d_in[p + 19];
    float* outP = (float*)d_out + (size_t)br * M * D;

    // x -> bf16
    k_f32_to_bf16<<<2048, 256, 0, stream>>>(xin, xbf, M * D);

    // GEMM1: g0 = x @ Wg + bg          (bf16 out)
    k_transpose_bf16<<<dim3(D / 32, D / 32), tB, 0, stream>>>(Wg, wT, D, D);
    k_wmma_gemm<<<dim3(M / 256, D / 64), 256, 0, stream>>>(xbf, wT, bg, g0, nullptr, D, D, 0);

    // GEMM2: h = gelu(g0 @ W1 + b1)    (bf16 out)
    k_transpose_bf16<<<dim3(D / 32, D4 / 32), tB, 0, stream>>>(W1, wT, D, D4);
    k_wmma_gemm<<<dim3(M / 256, D4 / 64), 256, 0, stream>>>(g0, wT, b1, hbuf, nullptr, D4, D, 1);

    // GEMM3: gate = silu(h @ W2 + b2)  (f32 out)
    k_transpose_bf16<<<dim3(D4 / 32, D / 32), tB, 0, stream>>>(W2, wT, D4, D);
    k_wmma_gemm<<<dim3(M / 256, D / 64), 256, 0, stream>>>(hbuf, wT, b2, nullptr, gate, D, D4, 2);

    // GEMM4: val = x @ Wv + bv         (f32 out)
    k_transpose_bf16<<<dim3(D / 32, D / 32), tB, 0, stream>>>(Wv, wT, D, D);
    k_wmma_gemm<<<dim3(M / 256, D / 64), 256, 0, stream>>>(xbf, wT, bv, nullptr, valb, D, D, 0);

    // SSM coefficient precompute
    k_cb<<<D / 256, 256, 0, stream>>>(d1, B1, C1, CB1, D);
    k_cb<<<D / 256, 256, 0, stream>>>(d2, B2, C2, CB2, D);

    // fused conv + dual SSM + gating + residual
    k_conv_ssm_out<<<(4 * D) / 256, 256, 0, stream>>>(valb, gate, xin,
        cw1, cb1, cw2, cb2, A1, CB1, A2, CB2, outP, 4, 2048, D);
  }
}
